// SwitchFeedForwardTail_9929964389243
// MI455X (gfx1250) — matrix operand order repaired and checked by reading, not compiled, not run on hardware
//
#include <hip/hip_runtime.h>
#include <hip/hip_bf16.h>
#include <math.h>

#define D_MODEL   768
#define D_FF      3072
#define N_EXPERTS 9
#define NTOK      2048
#define TILE_M    32
#define FFCHUNK   128
#define XP        (D_MODEL + 8)   // LDS pitch (halves), rows stay 16B-aligned
#define HP        (FFCHUNK + 8)

// WMMA-tiled weight geometry: 32(K) x 16(N) f16 tiles, 512 halves = 1KB each
#define TILE_HALVES 512
#define W1_KT (D_MODEL / 32)      // 24
#define W1_NT (D_FF / 16)         // 192
#define W2_KT (D_FF / 32)         // 96
#define W2_NT (D_MODEL / 16)      // 48
#define W1_TILES (N_EXPERTS * W1_KT * W1_NT)   // 41472
#define W2_TILES (N_EXPERTS * W2_KT * W2_NT)   // 41472

typedef __attribute__((ext_vector_type(16))) _Float16 v16h;
typedef __attribute__((ext_vector_type(8)))  _Float16 v8h;
typedef __attribute__((ext_vector_type(4)))  _Float16 v4h;
typedef __attribute__((ext_vector_type(8)))  float    v8f;

// ---- A fragment (16x32 f16) from LDS, contiguous K order (k = 16*(lane>>4) + s) ----
__device__ __forceinline__ v16h load_a_frag(const _Float16* s, int pitch, int mt, int k0, int lane) {
  int m    = mt * 16 + (lane & 15);
  int koff = (lane >> 4) * 16;
  const _Float16* p = s + m * pitch + k0 + koff;
  v8h lo = *(const v8h*)(p);
  v8h hi = *(const v8h*)(p + 8);
  return __builtin_shufflevector(lo, hi, 0,1,2,3,4,5,6,7,8,9,10,11,12,13,14,15);
}

// ---- B fragment (32x16 f16) from pre-tiled weights: lane's 16 halves are
// contiguous at tile + lane*32B -> two global_load_b128 ------------------------
__device__ __forceinline__ v16h load_b_tile(const _Float16* __restrict__ tile, int lane) {
  const v8h* p = (const v8h*)tile + lane * 2;
  v8h lo = p[0];
  v8h hi = p[1];
  return __builtin_shufflevector(lo, hi, 0,1,2,3,4,5,6,7,8,9,10,11,12,13,14,15);
}

__device__ __forceinline__ float gelu_exact(float v) {
  return 0.5f * v * (1.0f + erff(v * 0.70710678118654752f));
}

// ---------------- 0a) x (fp32) -> xh (f16), row-major ------------------------
__global__ void __launch_bounds__(256) xh_kernel(const float* __restrict__ x,
                                                 _Float16* __restrict__ xh) {
  int i = blockIdx.x * 256 + threadIdx.x;       // one float4 per thread
  float4 v = ((const float4*)x)[i];
  v4h h;
  h[0] = (_Float16)v.x; h[1] = (_Float16)v.y;
  h[2] = (_Float16)v.z; h[3] = (_Float16)v.w;
  ((v4h*)xh)[i] = h;
}

// ---------------- 0b) W1/W2 (fp32) -> f16 WMMA-B-fragment-tiled --------------
// One wave per 32x16 tile. Lane l produces its own 16 halves
// (K = (l>>4)*16 + j, N = l&15) and writes them contiguously (1KB/wave, coalesced).
// W1T block order: (e, nt, kt)  -> GEMM1 K-walk is contiguous.
// W2T block order: (e, kt, nt)  -> GEMM2 N-walk per K-step is contiguous.
__global__ void __launch_bounds__(256) convert_w_kernel(
    const float* __restrict__ W1, const float* __restrict__ W2,
    _Float16* __restrict__ W1T, _Float16* __restrict__ W2T)
{
  int wid  = blockIdx.x * 8 + (threadIdx.x >> 5);
  int lane = threadIdx.x & 31;
  int nn   = lane & 15;
  int khi  = lane >> 4;
  const float* src;
  _Float16*    dst;
  int pitch;
  if (wid < W1_TILES) {
    int e  = wid / (W1_NT * W1_KT);
    int r  = wid % (W1_NT * W1_KT);
    int nt = r / W1_KT;
    int kt = r % W1_KT;
    src   = W1 + ((size_t)(e * D_MODEL + kt * 32 + khi * 16)) * D_FF + nt * 16 + nn;
    pitch = D_FF;
    dst   = W1T + (size_t)wid * TILE_HALVES + lane * 16;
  } else {
    int w2 = wid - W1_TILES;
    int e  = w2 / (W2_KT * W2_NT);
    int r  = w2 % (W2_KT * W2_NT);
    int kt = r / W2_NT;
    int nt = r % W2_NT;
    src   = W2 + ((size_t)(e * D_FF + kt * 32 + khi * 16)) * D_MODEL + nt * 16 + nn;
    pitch = D_MODEL;
    dst   = W2T + (size_t)w2 * TILE_HALVES + lane * 16;
  }
  v16h f;
#pragma unroll
  for (int j = 0; j < 16; ++j) f[j] = (_Float16)src[(size_t)j * pitch];
  *(v8h*)(dst)     = __builtin_shufflevector(f, f, 0,1,2,3,4,5,6,7);
  *(v8h*)(dst + 8) = __builtin_shufflevector(f, f, 8,9,10,11,12,13,14,15);
}

// ---------------- 1) Router ---------------------------------------------------
__global__ void __launch_bounds__(256) router_kernel(
    const float* __restrict__ x, const float* __restrict__ Ws,
    const float* __restrict__ bs, int* __restrict__ routes, float* __restrict__ pbuf)
{
  int t    = blockIdx.x * 8 + (threadIdx.x >> 5);
  int lane = threadIdx.x & 31;
  float acc[N_EXPERTS];
#pragma unroll
  for (int e = 0; e < N_EXPERTS; ++e) acc[e] = 0.0f;
  const float* xr = x + t * D_MODEL;
  for (int d = lane; d < D_MODEL; d += 32) {
    float xv = xr[d];
#pragma unroll
    for (int e = 0; e < N_EXPERTS; ++e) acc[e] += xv * Ws[d * N_EXPERTS + e];
  }
#pragma unroll
  for (int e = 0; e < N_EXPERTS; ++e)
    for (int off = 16; off > 0; off >>= 1)
      acc[e] += __shfl_xor(acc[e], off, 32);
  if (lane == 0) {
    float mx = -1e30f; int arg = 0;
#pragma unroll
    for (int e = 0; e < N_EXPERTS; ++e) {
      float l = acc[e] + bs[e];
      acc[e] = l;
      if (l > mx) { mx = l; arg = e; }
    }
    float s = 0.0f;
#pragma unroll
    for (int e = 0; e < N_EXPERTS; ++e) s += expf(acc[e] - mx);
    routes[t] = arg;
    pbuf[t]   = 1.0f / s;
  }
}

// ---------------- 2) Scatter --------------------------------------------------
__global__ void __launch_bounds__(1024) scatter_kernel(
    const int* __restrict__ routes, int* __restrict__ counts, int* __restrict__ lists)
{
  __shared__ int c[N_EXPERTS - 1];
  if (threadIdx.x < N_EXPERTS - 1) c[threadIdx.x] = 0;
  __syncthreads();
  for (int t = threadIdx.x; t < NTOK; t += 1024) {
    int r = routes[t];
    if (r > 0) {
      int pos = atomicAdd(&c[r - 1], 1);
      lists[(r - 1) * NTOK + pos] = t;
    }
  }
  __syncthreads();
  if (threadIdx.x < N_EXPERTS - 1) counts[threadIdx.x + 1] = c[threadIdx.x];
  if (threadIdx.x == N_EXPERTS - 1) counts[0] = NTOK;
}

// ---------------- 3) Expert FFN via WMMA --------------------------------------
__global__ void __launch_bounds__(256) moe_expert_kernel(
    const _Float16* __restrict__ xh,
    const _Float16* __restrict__ W1T, const float* __restrict__ b1,
    const _Float16* __restrict__ W2T, const float* __restrict__ b2,
    const int* __restrict__ counts, const int* __restrict__ lists,
    float* __restrict__ common, float* __restrict__ sel)
{
  const int e    = blockIdx.x;
  const int tile = blockIdx.y;
  const int cnt  = counts[e];
  if (tile * TILE_M >= cnt) return;       // uniform exit, no barriers crossed

  __shared__ _Float16 Xs[TILE_M * XP];
  __shared__ _Float16 Hs[TILE_M * HP];
  __shared__ int      toks[TILE_M];

  const int tid = threadIdx.x;
  if (tid < TILE_M) {
    int idx = tile * TILE_M + tid;
    int t   = -1;
    if (idx < cnt) t = (e == 0) ? idx : lists[(e - 1) * NTOK + idx];
    toks[tid] = t;
  }
  __syncthreads();

  {   // stage X tile: async global(f16 rows, gathered by token) -> LDS
    int row   = tid >> 3;          // 8 threads per row
    int cbase = (tid & 7) * 96;    // 96 halves = 12 x 16B per thread
    int t = toks[row]; if (t < 0) t = 0;
    const _Float16* gsrc = xh + (size_t)t * D_MODEL + cbase;
    unsigned int lds0 = (unsigned int)(size_t)(Xs + row * XP + cbase);
#pragma unroll
    for (int i = 0; i < 12; ++i) {
      unsigned long long ga = (unsigned long long)(size_t)(gsrc + i * 8);
      unsigned int       la = lds0 + i * 16;
      asm volatile("global_load_async_to_lds_b128 %0, %1, off"
                   :: "v"(la), "v"(ga) : "memory");
    }
    asm volatile("s_wait_asynccnt 0x0" ::: "memory");
  }
  __syncthreads();

  const int wave = tid >> 5;
  const int lane = tid & 31;
  const _Float16* W1Te = W1T + (size_t)e * (W1_NT * W1_KT) * TILE_HALVES;
  const _Float16* W2Te = W2T + (size_t)e * (W2_KT * W2_NT) * TILE_HALVES;
  const float* b1e = b1 + e * D_FF;
  const float* b2e = b2 + e * D_MODEL;

  v8f acc2[12] = {};   // persistent 32x768 output: 2 M-tiles x 6 N-tiles / wave

  for (int chunk = 0; chunk < D_FF / FFCHUNK; ++chunk) {
    const int ffbase = chunk * FFCHUNK;

    // ---- GEMM1: this wave computes H[:, ffbase + wave*16 .. +15]
    // B fragments double-buffered: prefetch K-tile k+1 before the wmma on k.
    v8f acc1a = {}, acc1b = {};
    const int nt1 = chunk * 8 + wave;                 // global N-tile in W1T
    const _Float16* bptr = W1Te + (size_t)nt1 * W1_KT * TILE_HALVES;
    v16h bcur = load_b_tile(bptr, lane);
    for (int k0 = 0; k0 < D_MODEL - 32; k0 += 32) {
      bptr += TILE_HALVES;
      v16h bnext = load_b_tile(bptr, lane);
      v16h a0 = load_a_frag(Xs, XP, 0, k0, lane);
      v16h a1 = load_a_frag(Xs, XP, 1, k0, lane);
      acc1a = __builtin_amdgcn_wmma_f32_16x16x32_f16(false, a0, false, bcur, (short)0, acc1a, false, false);
      acc1b = __builtin_amdgcn_wmma_f32_16x16x32_f16(false, a1, false, bcur, (short)0, acc1b, false, false);
      bcur = bnext;
    }
    {   // peeled last K-tile
      v16h a0 = load_a_frag(Xs, XP, 0, D_MODEL - 32, lane);
      v16h a1 = load_a_frag(Xs, XP, 1, D_MODEL - 32, lane);
      acc1a = __builtin_amdgcn_wmma_f32_16x16x32_f16(false, a0, false, bcur, (short)0, acc1a, false, false);
      acc1b = __builtin_amdgcn_wmma_f32_16x16x32_f16(false, a1, false, bcur, (short)0, acc1b, false, false);
    }
    // bias + exact GELU -> Hs (D layout: VGPR v -> row v / v+8 by lane half)
    {
      float bv = b1e[ffbase + wave * 16 + (lane & 15)];
      int ncol = wave * 16 + (lane & 15);
      int mhi  = (lane >> 4) << 3;
#pragma unroll
      for (int v = 0; v < 8; ++v) {
        Hs[(v + mhi) * HP + ncol]      = (_Float16)gelu_exact(acc1a[v] + bv);
        Hs[(16 + v + mhi) * HP + ncol] = (_Float16)gelu_exact(acc1b[v] + bv);
      }
    }
    __syncthreads();

    // ---- GEMM2: Out(32x768) += H_chunk(32x128) @ W2[ffbase..+127, :]
    // B fragments double-buffered across the 6-tile N-walk (condition folds
    // at compile time under full unroll).
    for (int k0 = 0; k0 < FFCHUNK; k0 += 32) {
      v16h a0 = load_a_frag(Hs, HP, 0, k0, lane);
      v16h a1 = load_a_frag(Hs, HP, 1, k0, lane);
      const _Float16* w2base =
          W2Te + ((size_t)((ffbase + k0) >> 5) * W2_NT + wave * 6) * TILE_HALVES;
      v16h bcur = load_b_tile(w2base, lane);
#pragma unroll
      for (int j = 0; j < 6; ++j) {
        v16h bnext = bcur;
        if (j < 5) bnext = load_b_tile(w2base + (size_t)(j + 1) * TILE_HALVES, lane);
        acc2[j]     = __builtin_amdgcn_wmma_f32_16x16x32_f16(false, a0, false, bcur, (short)0, acc2[j],     false, false);
        acc2[6 + j] = __builtin_amdgcn_wmma_f32_16x16x32_f16(false, a1, false, bcur, (short)0, acc2[6 + j], false, false);
        bcur = bnext;
      }
    }
    __syncthreads();   // WAR on Hs before next chunk
  }

  // ---- epilogue: + b2, scatter rows to common / sel
  float* dst = (e == 0) ? common : sel;
  int mhi = (lane >> 4) << 3;
#pragma unroll
  for (int j = 0; j < 6; ++j) {
    int n = (wave * 6 + j) * 16 + (lane & 15);
    float bias = b2e[n];
#pragma unroll
    for (int v = 0; v < 8; ++v) {
      int t0 = toks[v + mhi];
      if (t0 >= 0) dst[t0 * D_MODEL + n] = acc2[j][v] + bias;
      int t1 = toks[16 + v + mhi];
      if (t1 >= 0) dst[t1 * D_MODEL + n] = acc2[6 + j][v] + bias;
    }
  }
}

// ---------------- 4) Blend ----------------------------------------------------
__global__ void __launch_bounds__(256) blend_kernel(
    const float* __restrict__ common, const float* __restrict__ sel,
    const int* __restrict__ routes, const float* __restrict__ pbuf,
    float* __restrict__ out)
{
  int i = blockIdx.x * 256 + threadIdx.x;
  if (i >= NTOK * D_MODEL) return;
  int t   = i / D_MODEL;
  int r   = routes[t];
  float c = common[i];
  float p = pbuf[t];
  out[i]  = (r == 0) ? c : (p * sel[i] + (1.0f - p) * c);
}

// ---------------- host launcher -----------------------------------------------
extern "C" void kernel_launch(void* const* d_in, const int* in_sizes, int n_in,
                              void* d_out, int out_size, void* d_ws, size_t ws_size,
                              hipStream_t stream) {
  const float* x  = (const float*)d_in[0];
  const float* Ws = (const float*)d_in[1];
  const float* bs = (const float*)d_in[2];
  const float* W1 = (const float*)d_in[3];
  const float* b1 = (const float*)d_in[4];
  const float* W2 = (const float*)d_in[5];
  const float* b2 = (const float*)d_in[6];
  float* out = (float*)d_out;

  char* w = (char*)d_ws;
  int*      routes = (int*)w;       w += 8192;                        // 2048 ints
  float*    pbuf   = (float*)w;     w += 8192;                        // 2048 floats
  int*      counts = (int*)w;       w += 256;                         // 9 ints
  int*      lists  = (int*)w;       w += (N_EXPERTS - 1) * NTOK * 4;  // 64 KB
  float*    common = (float*)w;     w += (size_t)NTOK * D_MODEL * 4;  // 6 MB
  float*    sel    = (float*)w;     w += (size_t)NTOK * D_MODEL * 4;  // 6 MB
  _Float16* xh     = (_Float16*)w;  w += (size_t)NTOK * D_MODEL * 2;  // 3 MB
  _Float16* W1T    = (_Float16*)w;  w += (size_t)W1_TILES * TILE_HALVES * 2;  // 40.5 MB
  _Float16* W2T    = (_Float16*)w;  /* 40.5 MB */

  xh_kernel       <<<NTOK * D_MODEL / 4 / 256, 256, 0, stream>>>(x, xh);
  convert_w_kernel<<<(W1_TILES + W2_TILES) / 8, 256, 0, stream>>>(W1, W2, W1T, W2T);
  router_kernel   <<<NTOK / 8, 256, 0, stream>>>(x, Ws, bs, routes, pbuf);
  scatter_kernel  <<<1, 1024, 0, stream>>>(routes, counts, lists);
  moe_expert_kernel<<<dim3(N_EXPERTS, NTOK / TILE_M), 256, 0, stream>>>(
      xh, W1T, b1, W2T, b2, counts, lists, common, sel);
  blend_kernel    <<<(NTOK * D_MODEL + 255) / 256, 256, 0, stream>>>(
      common, sel, routes, pbuf, out);
}
